// SeqAttention_55327768708021
// MI455X (gfx1250) — matrix operand order repaired and checked by reading, not compiled, not run on hardware
//
#include <hip/hip_runtime.h>
#include <math.h>

// Problem constants (match reference)
#define BH_N   64
#define NH     8
#define MQ     1024
#define SPAN_L 1024
#define DH     64
#define KV_LEN (MQ + SPAN_L)

// Tiling
#define MT      32      // query rows per workgroup
#define XCHUNKS 66      // 16-wide banded-column chunks: x in [0, 1056)
#define SSTRIDE 1072    // LDS row stride (floats), padded
#define NWAVE   8
#define NTHREADS 256

typedef _Float16 v16h __attribute__((ext_vector_type(16)));
typedef float    v8f  __attribute__((ext_vector_type(8)));

static __device__ inline v8f wmma_f16(v16h a, v16h b, v8f c) {
  // D = A(16x32 f16) * B(32x16 f16) + C(16x16 f32)
  return __builtin_amdgcn_wmma_f32_16x16x32_f16(false, a, false, b, (short)0, c,
                                                false, false);
}

__global__ __launch_bounds__(NTHREADS, 1)
void adaptive_span_attn(const float* __restrict__ Q,
                        const float* __restrict__ K,
                        const float* __restrict__ V,
                        const float* __restrict__ PE,
                        const float* __restrict__ CV,
                        float* __restrict__ Out) {
  // Banded score/probability buffer: S[i][x], x = absolute_key - m0, x in [0,1056)
  __shared__ float S[MT][SSTRIDE];
  // PV partial-sum accumulation tile
  __shared__ float Oacc[MT][DH];

  const int m0   = blockIdx.x * MT;
  const int bh   = blockIdx.y;
  const int tid  = threadIdx.x;
  const int lane = tid & 31;
  const int wave = tid >> 5;
  const int ln   = lane & 15;   // WMMA column (N) index / A-row index
  const int lh   = lane >> 4;   // half-wave selector

  const float* Qb = Q + (size_t)bh * MQ * DH;
  const float* Kb = K + (size_t)bh * KV_LEN * DH;
  const float* Vb = V + (size_t)bh * KV_LEN * DH;
  float*       Ob = Out + (size_t)bh * MQ * DH;

  const float scale = 0.04419417382415922f;  // 1/sqrt(512), folded into Q

  // ---------------- Q A-fragments (register resident per wave) ----------------
  // A 16x32 f16 layout: lane row = lane&15; half lh: v<4 -> K=2v+8*lh, v>=4 -> K=16+2(v-4)+8*lh
  v16h qa[2][2];
  #pragma unroll
  for (int rt = 0; rt < 2; ++rt) {
    const float* qrow = Qb + (size_t)(m0 + 16 * rt + ln) * DH;
    #pragma unroll
    for (int kc = 0; kc < 2; ++kc) {
      v16h a;
      #pragma unroll
      for (int v = 0; v < 8; ++v) {
        int kb = ((v < 4) ? (2 * v) : (16 + 2 * (v - 4))) + 8 * lh + 32 * kc;
        a[2 * v]     = (_Float16)(qrow[kb] * scale);
        a[2 * v + 1] = (_Float16)(qrow[kb + 1] * scale);
      }
      qa[rt][kc] = a;
    }
  }

  // ---------------- Content scores: S[i][x] = q_i . k_{m0+x} ----------------
  // B 32x16 f16 layout: lane col = ln; v<4 -> K=2v+8*lh, v>=4 -> K=16+2(v-4)+8*lh (+1 in hi half of dword)
  for (int c = wave; c < XCHUNKS; c += NWAVE) {
    const float* krow = Kb + (size_t)(m0 + 16 * c + ln) * DH;
    v16h b0, b1;
    #pragma unroll
    for (int v = 0; v < 8; ++v) {
      int d = ((v < 4) ? (2 * v) : (16 + 2 * (v - 4))) + 8 * lh;
      b0[2 * v]     = (_Float16)krow[d];
      b0[2 * v + 1] = (_Float16)krow[d + 1];
      b1[2 * v]     = (_Float16)krow[32 + d];
      b1[2 * v + 1] = (_Float16)krow[32 + d + 1];
    }
    #pragma unroll
    for (int rt = 0; rt < 2; ++rt) {
      v8f acc = {};
      acc = wmma_f16(qa[rt][0], b0, acc);
      acc = wmma_f16(qa[rt][1], b1, acc);
      #pragma unroll
      for (int r = 0; r < 8; ++r)   // D: vgpr r -> row r + 8*lh, col ln
        S[16 * rt + r + 8 * lh][16 * c + ln] = acc[r];
    }
  }
  __syncthreads();

  // ---------------- Positional scores: S[i][i + l] += q_i . pe[:, l] ----------------
  for (int c = wave; c < 64; c += NWAVE) {
    const int l0 = 16 * c;
    v16h b0, b1;
    #pragma unroll
    for (int v = 0; v < 8; ++v) {
      int d = ((v < 4) ? (2 * v) : (16 + 2 * (v - 4))) + 8 * lh;
      b0[2 * v]     = (_Float16)PE[(size_t)d * SPAN_L + l0 + ln];
      b0[2 * v + 1] = (_Float16)PE[(size_t)(d + 1) * SPAN_L + l0 + ln];
      b1[2 * v]     = (_Float16)PE[(size_t)(32 + d) * SPAN_L + l0 + ln];
      b1[2 * v + 1] = (_Float16)PE[(size_t)(32 + d + 1) * SPAN_L + l0 + ln];
    }
    #pragma unroll
    for (int rt = 0; rt < 2; ++rt) {
      v8f acc = {};
      acc = wmma_f16(qa[rt][0], b0, acc);
      acc = wmma_f16(qa[rt][1], b1, acc);
      #pragma unroll
      for (int r = 0; r < 8; ++r) {
        int i = 16 * rt + r + 8 * lh;
        S[i][i + l0 + ln] += acc[r];   // unique (i, x) per lane -> race-free
      }
    }
  }
  __syncthreads();

  // ---------------- Softmax + adaptive-span mask + renorm (in place) ----------------
  {
    const int i  = tid >> 3;          // row 0..31 (4 rows per wave, 8 lanes/row)
    const int s8 = tid & 7;
    const float cv = CV[bh & (NH - 1)];   // head = bh % NH
    float* Srow = S[i];

    float mx = -3.0e38f;
    for (int k = 0; k < 128; ++k) mx = fmaxf(mx, Srow[i + s8 + 8 * k]);
    mx = fmaxf(mx, __shfl_xor(mx, 1, 32));
    mx = fmaxf(mx, __shfl_xor(mx, 2, 32));
    mx = fmaxf(mx, __shfl_xor(mx, 4, 32));

    float Zs = 0.f, Zm = 0.f;
    for (int k = 0; k < 128; ++k) {
      int l = s8 + 8 * k;
      float p = __expf(Srow[i + l] - mx);
      // mask = clip(((l - 1023) + cv*SPAN)/RAMP + 1, 0, 1)
      float msk = fminf(fmaxf(((float)l - 1023.0f + cv * (float)SPAN_L) * (1.0f / 32.0f) + 1.0f,
                              0.0f), 1.0f);
      float pm = p * msk;
      Zs += p;
      Zm += pm;
      Srow[i + l] = pm;
    }
    Zs += __shfl_xor(Zs, 1, 32); Zs += __shfl_xor(Zs, 2, 32); Zs += __shfl_xor(Zs, 4, 32);
    Zm += __shfl_xor(Zm, 1, 32); Zm += __shfl_xor(Zm, 2, 32); Zm += __shfl_xor(Zm, 4, 32);
    // attn = softmax*mask / (sum(softmax*mask) + 1e-8)  ==  pm / (Zm + 1e-8*Zs)
    const float rden = 1.0f / (Zm + 1e-8f * Zs);
    for (int k = 0; k < 128; ++k) Srow[i + s8 + 8 * k] *= rden;

    // zero the 32 out-of-band cells of this row: x<i and i+1024<=x<1056
    #pragma unroll
    for (int zz = 0; zz < 4; ++zz) {
      int z = 4 * s8 + zz;
      int x = (z < i) ? z : (1024 + z);
      Srow[x] = 0.0f;
    }
  }
  __syncthreads();

  // ---------------- PV: Out[i][d] = sum_x P[i][x] * V[m0+x][d] ----------------
  // k-split: each wave keeps all 8 output tiles (2 row-tiles x 4 d-chunks) and
  // processes k-chunks kkc = wave, wave+8, ...  A built once per row-tile and
  // reused across 4 d-chunks; B built once per d-chunk and reused across both
  // row-tiles -> ~12 loads + 6 cvts per WMMA instead of 32 + 16.
  {
    // zero the accumulation tile (combined across waves via ds_add_f32)
    #pragma unroll
    for (int e = 0; e < (MT * DH) / NTHREADS; ++e)
      (&Oacc[0][0])[tid * ((MT * DH) / NTHREADS) + e] = 0.0f;

    v8f acc[2][4];
    #pragma unroll
    for (int rt = 0; rt < 2; ++rt)
      #pragma unroll
      for (int dc = 0; dc < 4; ++dc)
        acc[rt][dc] = (v8f){};

    for (int kkc = wave; kkc < 33; kkc += NWAVE) {   // 33 * 32 = 1056 banded cols
      v16h a[2];
      #pragma unroll
      for (int rt = 0; rt < 2; ++rt) {
        const float* srow = &S[16 * rt + ln][32 * kkc];
        #pragma unroll
        for (int v = 0; v < 8; ++v) {
          int kb = ((v < 4) ? (2 * v) : (16 + 2 * (v - 4))) + 8 * lh;
          a[rt][2 * v]     = (_Float16)srow[kb];
          a[rt][2 * v + 1] = (_Float16)srow[kb + 1];
        }
      }
      #pragma unroll
      for (int dc = 0; dc < 4; ++dc) {
        v16h b;
        #pragma unroll
        for (int v = 0; v < 8; ++v) {
          int r = ((v < 4) ? (2 * v) : (16 + 2 * (v - 4))) + 8 * lh;
          const float* vrow = Vb + (size_t)(m0 + 32 * kkc + r) * DH + 16 * dc + ln;
          b[2 * v]     = (_Float16)vrow[0];
          b[2 * v + 1] = (_Float16)vrow[DH];
        }
        acc[0][dc] = wmma_f16(a[0], b, acc[0][dc]);
        acc[1][dc] = wmma_f16(a[1], b, acc[1][dc]);
      }
    }

    __syncthreads();   // all zeroing + all partials computed before accumulation
    #pragma unroll
    for (int rt = 0; rt < 2; ++rt)
      #pragma unroll
      for (int dc = 0; dc < 4; ++dc)
        #pragma unroll
        for (int r = 0; r < 8; ++r)
          atomicAdd(&Oacc[16 * rt + r + 8 * lh][16 * dc + ln], acc[rt][dc][r]);
    __syncthreads();

    // coalesced store of the 32x64 output tile
    #pragma unroll
    for (int e = 0; e < (MT * DH) / NTHREADS; ++e) {
      int idx = tid * ((MT * DH) / NTHREADS) + e;
      Ob[(size_t)(m0 + (idx >> 6)) * DH + (idx & 63)] = (&Oacc[0][0])[idx];
    }
  }
}

extern "C" void kernel_launch(void* const* d_in, const int* in_sizes, int n_in,
                              void* d_out, int out_size, void* d_ws, size_t ws_size,
                              hipStream_t stream) {
  (void)in_sizes; (void)n_in; (void)out_size; (void)d_ws; (void)ws_size;
  const float* q  = (const float*)d_in[0];
  const float* k  = (const float*)d_in[1];
  const float* v  = (const float*)d_in[2];
  const float* pe = (const float*)d_in[3];
  const float* cv = (const float*)d_in[4];
  float* out = (float*)d_out;

  dim3 grid(MQ / MT, BH_N);
  adaptive_span_attn<<<grid, NTHREADS, 0, stream>>>(q, k, v, pe, cv, out);
}
